// mtl_net_25288767439363
// MI455X (gfx1250) — matrix operand order repaired and checked by reading, not compiled, not run on hardware
//
#include <hip/hip_runtime.h>
#include <stdint.h>

// ---- problem dims ----
#define B_   32
#define T_   128
#define DIN  32
#define H1_  1024
#define H_   2048
#define PEN_ 1024
#define C_   64
#define P_   4
#define G3H  (3 * H_)   // 6144
#define BT   (B_ * T_)  // 4096
#define GRU_BLOCKS (H_ / 16)  // 128
#define BK   256              // K-chunk staged to LDS by the TDM

typedef __attribute__((ext_vector_type(16))) __bf16 v16bf;
typedef __attribute__((ext_vector_type(8)))  float  v8f;
typedef __attribute__((ext_vector_type(4)))  unsigned int v4u_t;
typedef __attribute__((ext_vector_type(8)))  int v8i_t;
typedef __attribute__((ext_vector_type(4)))  int v4i_t;
typedef unsigned short bf16_t;

__device__ __forceinline__ bf16_t f2bf(float f) {
  unsigned int u = __float_as_uint(f);
  u += 0x7FFFu + ((u >> 16) & 1u);   // round-to-nearest-even
  return (bf16_t)(u >> 16);
}

union FragB16 { v16bf v; unsigned int u[8]; };
union FragF   { v8f   v; float        f[8]; };

// A fragment: 16x32 bf16 tile (row-major, stride ldk). CDNA5 layout:
// lanes 0-15 hold rows 0-15 with K {0..7,16..23}; lanes 16-31 same rows, K {8..15,24..31}.
// Works for both global and LDS source pointers.
__device__ __forceinline__ v16bf load_frag_a(const bf16_t* __restrict__ A, int ldk) {
  const int lane = threadIdx.x & 31;
  const int row  = lane & 15;
  const int kh   = (lane >> 4) * 8;
  FragB16 f;
  const bf16_t* rp = A + (size_t)row * ldk;
#pragma unroll
  for (int d = 0; d < 8; ++d) {
    int k = (d < 4) ? (kh + 2 * d) : (16 + kh + 2 * (d - 4));
    f.u[d] = *(const unsigned int*)(rp + k);
  }
  return f.v;
}

// B fragment: 32x16 tile of W^T, i.e. B[k][n] = W[n][k] with W row-major (stride ldw).
__device__ __forceinline__ v16bf load_frag_b(const bf16_t* __restrict__ W, int ldw) {
  const int lane = threadIdx.x & 31;
  const int n  = lane & 15;
  const int kh = (lane >> 4) * 8;
  FragB16 f;
  const bf16_t* rp = W + (size_t)n * ldw + kh;
#pragma unroll
  for (int d = 0; d < 8; ++d) f.u[d] = *(const unsigned int*)(rp + ((d < 4) ? 2 * d : 16 + 2 * (d - 4)));
  return f.v;
}

__device__ __forceinline__ v8f wmma_bf16(v16bf a, v16bf b, v8f c) {
  return __builtin_amdgcn_wmma_f32_16x16x32_bf16(false, a, false, b, (short)0, c, false, false);
}

// ---- Tensor Data Mover: stage a rows x cw bf16 tile (row stride ldk_global) to LDS ----
// D# built per CDNA5 ISA 8.3/8.4 (2D tensor, data_size=2B, no pad/iterate/multicast).
// Toolchain uses the 6-arg builtin: (g0 v4u, g1 v8i, g2 v4i, g3 v4i, g4 v8i, cpol).
__device__ __forceinline__ void tdm_load_a(unsigned int lds_off, const bf16_t* gaddr,
                                           int rows, int cw, int ldk_global) {
  unsigned long long ga = (unsigned long long)(uintptr_t)gaddr;
  v4u_t g0;
  g0[0] = 1u;                                   // count=1, user descriptor
  g0[1] = lds_off;                              // lds_addr (bytes)
  g0[2] = (unsigned int)ga;                     // global_addr[31:0]
  g0[3] = (unsigned int)(ga >> 32) | (2u << 30);// global_addr[56:32] | type=2
  unsigned int td0 = (unsigned int)cw;          // tensor_dim0 (elements)
  unsigned int td1 = (unsigned int)rows;        // tensor_dim1
  unsigned long long s0 = (unsigned long long)(unsigned int)ldk_global; // dim0 stride
  v8i_t g1;
  g1[0] = (int)(1u << 16);                                  // data_size=1 -> 2 bytes
  g1[1] = (int)((td0 & 0xFFFFu) << 16);                     // tensor_dim0 lo16
  g1[2] = (int)((td0 >> 16) | ((td1 & 0xFFFFu) << 16));     // td0 hi16 | td1 lo16
  g1[3] = (int)((td1 >> 16) | (((unsigned int)cw) << 16));  // td1 hi16 | tile_dim0
  g1[4] = (int)(rows & 0xFFFF);                             // tile_dim1 | tile_dim2=0
  g1[5] = (int)(unsigned int)s0;                            // dim0_stride lo32
  g1[6] = (int)(unsigned int)(s0 >> 32);                    // dim0_stride hi16
  g1[7] = 0;
  v4i_t g2 = {0, 0, 0, 0};
  v4i_t g3 = {0, 0, 0, 0};
  v8i_t g4 = {0, 0, 0, 0, 0, 0, 0, 0};
  __builtin_amdgcn_tensor_load_to_lds(g0, g1, g2, g3, g4, 0);
}

// ---------------- float -> bf16 conversion ----------------
__global__ void cvt_kernel(const float* __restrict__ s, bf16_t* __restrict__ d, int n) {
  for (int i = blockIdx.x * blockDim.x + threadIdx.x; i < n; i += gridDim.x * blockDim.x)
    d[i] = f2bf(s[i]);
}

// ---------------- zero-init h0 + barrier counter ----------------
__global__ void init_h_kernel(float* __restrict__ hf, bf16_t* __restrict__ hb, int n,
                              unsigned int* __restrict__ bar) {
  int i0 = blockIdx.x * blockDim.x + threadIdx.x;
  if (i0 == 0) *bar = 0u;
  for (int i = i0; i < n; i += gridDim.x * blockDim.x) {
    hf[i] = 0.f;
    hb[i] = 0;
  }
}

// ---------------- generic bf16 WMMA GEMM: C[M,N] = A[M,K] * W[N,K]^T + bias ----------------
// one wave per 32x64 output tile; block = 8 waves -> 512 cols, all sharing one 32-row A slab.
// A slab is TDM-staged into LDS (double-buffered); B streams from global (L2-resident).
__global__ void gemm_bf16_kernel(const bf16_t* __restrict__ A, const bf16_t* __restrict__ W,
                                 const float* __restrict__ bias,
                                 float* __restrict__ Cf, bf16_t* __restrict__ Cb,
                                 int M, int N, int K, int relu) {
  __shared__ bf16_t As[2][32 * BK];
  const int wave = threadIdx.x >> 5;
  const int lane = threadIdx.x & 31;
  const int n0 = blockIdx.x * 512 + wave * 64;
  const int m0 = blockIdx.y * 32;
  const v8f z = {0.f, 0.f, 0.f, 0.f, 0.f, 0.f, 0.f, 0.f};
  v8f acc[2][4];
#pragma unroll
  for (int mf = 0; mf < 2; ++mf)
#pragma unroll
    for (int nf = 0; nf < 4; ++nf) acc[mf][nf] = z;

  const bf16_t* Wp[4];
#pragma unroll
  for (int nf = 0; nf < 4; ++nf) Wp[nf] = W + (size_t)(n0 + 16 * nf) * K;

  const int cw = (K < BK) ? K : BK;   // chunk width (uniform: all K are 32 or multiples of BK)
  const int nk = K / cw;

  // prologue: stage chunk 0
  if (wave == 0) {
    tdm_load_a((unsigned int)(uintptr_t)&As[0][0], A + (size_t)m0 * K, 32, cw, K);
    __builtin_amdgcn_s_wait_tensorcnt(0);
  }
  __syncthreads();

  for (int kc = 0; kc < nk; ++kc) {
    if (wave == 0 && kc + 1 < nk)   // async: stage next chunk while computing this one
      tdm_load_a((unsigned int)(uintptr_t)&As[(kc + 1) & 1][0],
                 A + (size_t)m0 * K + (size_t)(kc + 1) * cw, 32, cw, K);

    const bf16_t* Asl = &As[kc & 1][0];
    const int kg0 = kc * cw;
    for (int kk = 0; kk < cw; kk += 32) {
      const int kg = kg0 + kk;
      if (kg + 32 < K) {  // gfx1250 global_prefetch path for the weight streams
        __builtin_prefetch(Wp[0] + kg + 32, 0, 1);
        __builtin_prefetch(Wp[2] + kg + 32, 0, 1);
      }
      // issue all tile loads first, then the 8-WMMA burst
      v16bf b0 = load_frag_b(Wp[0] + kg, K);
      v16bf b1 = load_frag_b(Wp[1] + kg, K);
      v16bf b2 = load_frag_b(Wp[2] + kg, K);
      v16bf b3 = load_frag_b(Wp[3] + kg, K);
      v16bf a0 = load_frag_a(Asl + kk, cw);
      v16bf a1 = load_frag_a(Asl + 16 * cw + kk, cw);
      acc[0][0] = wmma_bf16(a0, b0, acc[0][0]);
      acc[1][0] = wmma_bf16(a1, b0, acc[1][0]);
      acc[0][1] = wmma_bf16(a0, b1, acc[0][1]);
      acc[1][1] = wmma_bf16(a1, b1, acc[1][1]);
      acc[0][2] = wmma_bf16(a0, b2, acc[0][2]);
      acc[1][2] = wmma_bf16(a1, b2, acc[1][2]);
      acc[0][3] = wmma_bf16(a0, b3, acc[0][3]);
      acc[1][3] = wmma_bf16(a1, b3, acc[1][3]);
    }

    if (kc + 1 < nk) {
      if (wave == 0) __builtin_amdgcn_s_wait_tensorcnt(0);
      __syncthreads();
    }
  }

#pragma unroll
  for (int mf = 0; mf < 2; ++mf)
#pragma unroll
    for (int nf = 0; nf < 4; ++nf) {
      FragF fr; fr.v = acc[mf][nf];
#pragma unroll
      for (int i = 0; i < 8; ++i) {
        int m = m0 + mf * 16 + ((lane >> 4) * 8) + i;
        int n = n0 + nf * 16 + (lane & 15);
        float v = fr.f[i] + bias[n];
        if (relu) v = fmaxf(v, 0.f);
        if (Cf) Cf[(size_t)m * N + n] = v;
        if (Cb) Cb[(size_t)m * N + n] = f2bf(v);
      }
    }
}

// ---------------- persistent GRU: all 128 timesteps in one launch ----------------
__global__ void gru_persistent_kernel(bf16_t* __restrict__ hb0, float* __restrict__ hf0,
                                      bf16_t* __restrict__ hb1, float* __restrict__ hf1,
                                      const bf16_t* __restrict__ Whh,
                                      const float* __restrict__ bhh,
                                      const float* __restrict__ XP,
                                      bf16_t* __restrict__ HS,
                                      unsigned int* __restrict__ bar) {
  const int lane = threadIdx.x & 31;
  const int n0 = blockIdx.x * 16;
  const int col = n0 + (lane & 15);
  const float br = bhh[col];
  const float bz = bhh[H_ + col];
  const float bn = bhh[2 * H_ + col];
  const v8f z = {0.f, 0.f, 0.f, 0.f, 0.f, 0.f, 0.f, 0.f};
  const bf16_t* Wg0 = Whh + (size_t)(0 * H_ + n0) * H_;
  const bf16_t* Wg1 = Whh + (size_t)(1 * H_ + n0) * H_;
  const bf16_t* Wg2 = Whh + (size_t)(2 * H_ + n0) * H_;

  for (int t = 0; t < T_; ++t) {
    const bf16_t* hb   = (t & 1) ? hb1 : hb0;
    const float*  hf   = (t & 1) ? hf1 : hf0;
    bf16_t*       hb_o = (t & 1) ? hb0 : hb1;
    float*        hf_o = (t & 1) ? hf0 : hf1;

    v8f acc[3][2];
#pragma unroll
    for (int g = 0; g < 3; ++g) { acc[g][0] = z; acc[g][1] = z; }

    for (int k = 0; k < H_; k += 32) {
      v16bf bg0 = load_frag_b(Wg0 + k, H_);
      v16bf bg1 = load_frag_b(Wg1 + k, H_);
      v16bf bg2 = load_frag_b(Wg2 + k, H_);
      v16bf a0  = load_frag_a(hb + k, H_);
      v16bf a1  = load_frag_a(hb + (size_t)16 * H_ + k, H_);
      acc[0][0] = wmma_bf16(a0, bg0, acc[0][0]);
      acc[0][1] = wmma_bf16(a1, bg0, acc[0][1]);
      acc[1][0] = wmma_bf16(a0, bg1, acc[1][0]);
      acc[1][1] = wmma_bf16(a1, bg1, acc[1][1]);
      acc[2][0] = wmma_bf16(a0, bg2, acc[2][0]);
      acc[2][1] = wmma_bf16(a1, bg2, acc[2][1]);
    }

#pragma unroll
    for (int mf = 0; mf < 2; ++mf) {
      FragF f0, f1, f2;
      f0.v = acc[0][mf]; f1.v = acc[1][mf]; f2.v = acc[2][mf];
#pragma unroll
      for (int i = 0; i < 8; ++i) {
        int b = mf * 16 + ((lane >> 4) * 8) + i;
        float hr = f0.f[i] + br;
        float hz = f1.f[i] + bz;
        float hn = f2.f[i] + bn;
        const float* xpr = XP + (size_t)(b * T_ + t) * G3H;
        float xr = xpr[col], xz = xpr[H_ + col], xn = xpr[2 * H_ + col];
        float r  = 1.f / (1.f + __expf(-(xr + hr)));
        float zz = 1.f / (1.f + __expf(-(xz + hz)));
        float nn = tanhf(xn + r * hn);
        float hnew = (1.f - zz) * nn + zz * hf[(size_t)b * H_ + col];
        hf_o[(size_t)b * H_ + col] = hnew;
        bf16_t hv = f2bf(hnew);
        hb_o[(size_t)b * H_ + col] = hv;
        HS[(size_t)(b * T_ + t) * H_ + col] = hv;
      }
    }

    if (t + 1 < T_) {  // device-scope barrier between timesteps
      __threadfence();
      if (lane == 0) {
        __hip_atomic_fetch_add(bar, 1u, __ATOMIC_RELEASE, __HIP_MEMORY_SCOPE_AGENT);
        const unsigned int target = (unsigned int)(GRU_BLOCKS * (t + 1));
        while (__hip_atomic_load(bar, __ATOMIC_ACQUIRE, __HIP_MEMORY_SCOPE_AGENT) < target)
          __builtin_amdgcn_s_sleep(1);
      }
      __threadfence();
    }
  }
}

// ---------------- label-gathered heads ----------------
__device__ float block_reduce(float v, float* red) {
  int tid = threadIdx.x;
  __syncthreads();
  red[tid] = v;
  __syncthreads();
  for (int s = 128; s > 0; s >>= 1) {
    if (tid < s) red[tid] += red[tid + s];
    __syncthreads();
  }
  return red[0];
}

__global__ void heads_kernel(const float* __restrict__ S, const int* __restrict__ labels,
                             const float* __restrict__ W4, const float* __restrict__ b4,
                             const float* __restrict__ W5, const float* __restrict__ b5,
                             const float* __restrict__ W6, const float* __restrict__ b6,
                             const float* __restrict__ Wp, const float* __restrict__ bp,
                             float* __restrict__ out) {
  const int bt = blockIdx.x;
  const int b = bt >> 7, t = bt & 127;
  const int lab = labels[b];
  const int tid = threadIdx.x;
  const float* s  = S  + (size_t)bt  * PEN_;
  const float* w4 = W4 + (size_t)lab * PEN_;
  const float* w5 = W5 + (size_t)lab * PEN_;
  const float* w6 = W6 + (size_t)lab * PEN_;
  float a4 = 0.f, a5 = 0.f, a6 = 0.f, ap[P_] = {0.f, 0.f, 0.f, 0.f};
  for (int d = tid; d < PEN_; d += 256) {
    float sv = s[d];
    a4 += sv * w4[d];
    a5 += sv * w5[d];
    a6 += sv * w6[d];
#pragma unroll
    for (int p = 0; p < P_; ++p)
      ap[p] += sv * Wp[((size_t)(p * C_ + lab)) * PEN_ + d];
  }
  __shared__ float red[256];
  a4 = block_reduce(a4, red);
  a5 = block_reduce(a5, red);
  a6 = block_reduce(a6, red);
#pragma unroll
  for (int p = 0; p < P_; ++p) ap[p] = block_reduce(ap[p], red);

  if (tid == 0) {
    out[bt]          = a4 + b4[lab];
    out[BT + bt]     = a5 + b5[lab];
    out[2 * BT + bt] = a6 + b6[lab];
#pragma unroll
    for (int p = 0; p < P_; ++p) {
      float v = ap[p] + bp[p * C_ + lab];
      out[3 * BT + (size_t)(p * B_ + b) * T_ + t] = 1.f / (1.f + __expf(-v));
    }
    if (bt == 0) out[3 * BT + P_ * BT] = 0.f;  // trailing scalar 0
  }
}

// ---------------- host ----------------
static inline void* ws_alloc(char* ws, size_t& off, size_t bytes) {
  void* p = ws + off;
  off = (off + bytes + 255) & ~(size_t)255;
  return p;
}

extern "C" void kernel_launch(void* const* d_in, const int* in_sizes, int n_in,
                              void* d_out, int out_size, void* d_ws, size_t ws_size,
                              hipStream_t stream) {
  const float* x    = (const float*)d_in[0];
  const int*   lab  = (const int*)  d_in[1];
  const float* W1   = (const float*)d_in[2];
  const float* b1   = (const float*)d_in[3];
  const float* W2   = (const float*)d_in[4];
  const float* b2   = (const float*)d_in[5];
  const float* W_ih = (const float*)d_in[6];
  const float* b_ih = (const float*)d_in[7];
  const float* W_hh = (const float*)d_in[8];
  const float* b_hh = (const float*)d_in[9];
  const float* W3   = (const float*)d_in[10];
  const float* b3   = (const float*)d_in[11];
  const float* W4   = (const float*)d_in[12];
  const float* b4   = (const float*)d_in[13];
  const float* W5   = (const float*)d_in[14];
  const float* b5   = (const float*)d_in[15];
  const float* W6   = (const float*)d_in[16];
  const float* b6   = (const float*)d_in[17];
  const float* Wp   = (const float*)d_in[18];
  const float* bp   = (const float*)d_in[19];
  float* out = (float*)d_out;

  char* ws = (char*)d_ws;
  size_t off = 0;
  bf16_t* xb   = (bf16_t*)ws_alloc(ws, off, (size_t)BT * DIN * 2);
  bf16_t* W1b  = (bf16_t*)ws_alloc(ws, off, (size_t)H1_ * DIN * 2);
  bf16_t* W2b  = (bf16_t*)ws_alloc(ws, off, (size_t)H_ * H1_ * 2);
  bf16_t* Wihb = (bf16_t*)ws_alloc(ws, off, (size_t)G3H * H_ * 2);
  bf16_t* Whhb = (bf16_t*)ws_alloc(ws, off, (size_t)G3H * H_ * 2);
  bf16_t* W3b  = (bf16_t*)ws_alloc(ws, off, (size_t)PEN_ * H_ * 2);
  bf16_t* A1b  = (bf16_t*)ws_alloc(ws, off, (size_t)BT * H1_ * 2);
  bf16_t* A2b  = (bf16_t*)ws_alloc(ws, off, (size_t)BT * H_ * 2);
  float*  XP   = (float*) ws_alloc(ws, off, (size_t)BT * G3H * 4);
  bf16_t* HS   = (bf16_t*)ws_alloc(ws, off, (size_t)BT * H_ * 2);
  float*  Sbuf = (float*) ws_alloc(ws, off, (size_t)BT * PEN_ * 4);
  float*  hf[2]; bf16_t* hb[2];
  hf[0] = (float*) ws_alloc(ws, off, (size_t)B_ * H_ * 4);
  hf[1] = (float*) ws_alloc(ws, off, (size_t)B_ * H_ * 4);
  hb[0] = (bf16_t*)ws_alloc(ws, off, (size_t)B_ * H_ * 2);
  hb[1] = (bf16_t*)ws_alloc(ws, off, (size_t)B_ * H_ * 2);
  unsigned int* bar = (unsigned int*)ws_alloc(ws, off, 256);

  auto cvt = [&](const float* s, bf16_t* d, int n) {
    int g = (n + 255) / 256; if (g > 4096) g = 4096;
    cvt_kernel<<<g, 256, 0, stream>>>(s, d, n);
  };
  cvt(x, xb, BT * DIN);
  cvt(W1, W1b, H1_ * DIN);
  cvt(W2, W2b, H_ * H1_);
  cvt(W_ih, Wihb, G3H * H_);
  cvt(W_hh, Whhb, G3H * H_);
  cvt(W3, W3b, PEN_ * H_);

  // layer 1: relu(x @ W1^T + b1) -> A1b (bf16)
  gemm_bf16_kernel<<<dim3(H1_ / 512, BT / 32), 256, 0, stream>>>(
      xb, W1b, b1, nullptr, A1b, BT, H1_, DIN, 1);
  // layer 2: relu(A1 @ W2^T + b2) -> A2b (bf16)
  gemm_bf16_kernel<<<dim3(H_ / 512, BT / 32), 256, 0, stream>>>(
      A1b, W2b, b2, nullptr, A2b, BT, H_, H1_, 1);
  // GRU input projection: XP = A2 @ W_ih^T + b_ih (f32)
  gemm_bf16_kernel<<<dim3(G3H / 512, BT / 32), 256, 0, stream>>>(
      A2b, Wihb, b_ih, XP, nullptr, BT, G3H, H_, 0);

  // h0 = 0, barrier counter = 0
  init_h_kernel<<<256, 256, 0, stream>>>(hf[0], hb[0], B_ * H_, bar);

  // all 128 GRU timesteps in one persistent launch
  gru_persistent_kernel<<<GRU_BLOCKS, 32, 0, stream>>>(
      hb[0], hf[0], hb[1], hf[1], Whhb, b_hh, XP, HS, bar);

  // out_s = relu(HS @ W3^T + b3) -> Sbuf (f32)
  gemm_bf16_kernel<<<dim3(PEN_ / 512, BT / 32), 256, 0, stream>>>(
      HS, W3b, b3, Sbuf, nullptr, BT, PEN_, H_, 1);

  // heads
  heads_kernel<<<BT, 256, 0, stream>>>(Sbuf, lab, W4, b4, W5, b5, W6, b6, Wp, bp, out);
  (void)in_sizes; (void)n_in; (void)out_size; (void)ws_size;
}
